// MLA_76647986364968
// MI455X (gfx1250) — matrix operand order repaired and checked by reading, not compiled, not run on hardware
//
#include <hip/hip_runtime.h>

// ---------------- types ----------------
typedef __bf16 bf16_t;
typedef bf16_t v16bf __attribute__((ext_vector_type(16)));
typedef bf16_t v8bf  __attribute__((ext_vector_type(8)));
typedef bf16_t v4bf  __attribute__((ext_vector_type(4)));
typedef float  v8f   __attribute__((ext_vector_type(8)));
typedef float  v4f   __attribute__((ext_vector_type(4)));
typedef int    i32x4 __attribute__((ext_vector_type(4)));

// ---------------- async global->LDS (CDNA5) ----------------
#if defined(__has_builtin)
# if __has_builtin(__builtin_amdgcn_global_load_async_to_lds_b128)
#  define HAS_ASYNC_LDS 1
# endif
#endif
#ifndef HAS_ASYNC_LDS
# define HAS_ASYNC_LDS 0
#endif

__device__ inline void async_cp16(const bf16_t* g, bf16_t* l) {
#if HAS_ASYNC_LDS
    __builtin_amdgcn_global_load_async_to_lds_b128(
        (i32x4*)g, (i32x4*)l, 0, 0);
#else
    *(v8bf*)l = *(const v8bf*)g;
#endif
}

__device__ inline void wait_async() {
#if HAS_ASYNC_LDS
# if defined(__has_builtin) && __has_builtin(__builtin_amdgcn_s_wait_asynccnt)
    __builtin_amdgcn_s_wait_asynccnt(0);
# else
    asm volatile("s_wait_asynccnt 0x0" ::: "memory");
# endif
#endif
}

// ---------------- problem constants ----------------
constexpr int B_   = 2;
constexpr int S_   = 2048;
constexpr int DIM_ = 2048;
constexpr int H_   = 16;
constexpr int NOPE_ = 128;
constexpr int ROPE_ = 64;
constexpr int QKHD_ = NOPE_ + ROPE_;   // 192
constexpr int VHD_  = 128;
constexpr int KV_RANK_ = 512;
constexpr int M_  = B_ * S_;           // 4096

// ---------------- workspace layout (bytes) ----------------
constexpr size_t SZ_QF   = (size_t)M_ * (H_*QKHD_) * 4;
constexpr size_t SZ_KVF  = (size_t)M_ * (KV_RANK_ + H_*ROPE_) * 4;
constexpr size_t SZ_KNF  = (size_t)M_ * (H_*NOPE_) * 4;
constexpr size_t SZ_VF   = (size_t)M_ * (H_*VHD_) * 4;
constexpr size_t SZ_XB   = (size_t)M_ * DIM_ * 2;
constexpr size_t SZ_WQT  = (size_t)(H_*QKHD_) * DIM_ * 2;
constexpr size_t SZ_WKVT = (size_t)(KV_RANK_+H_*ROPE_) * DIM_ * 2;
constexpr size_t SZ_WKT  = (size_t)(H_*NOPE_) * KV_RANK_ * 2;
constexpr size_t SZ_WVT  = (size_t)(H_*VHD_) * KV_RANK_ * 2;
constexpr size_t SZ_WOT  = (size_t)DIM_ * (H_*VHD_) * 2;
constexpr size_t SZ_CB   = (size_t)M_ * KV_RANK_ * 2;
constexpr size_t SZ_QB   = (size_t)M_ * (H_*QKHD_) * 2;
constexpr size_t SZ_KB   = SZ_QB;
constexpr size_t SZ_VT   = (size_t)M_ * (H_*VHD_) * 2;

constexpr size_t OFF_QF   = 0;
constexpr size_t OFF_KVF  = OFF_QF   + SZ_QF;
constexpr size_t OFF_KNF  = OFF_KVF  + SZ_KVF;
constexpr size_t OFF_VF   = OFF_KNF  + SZ_KNF;
constexpr size_t OFF_XB   = OFF_VF   + SZ_VF;
constexpr size_t OFF_WQT  = OFF_XB   + SZ_XB;
constexpr size_t OFF_WKVT = OFF_WQT  + SZ_WQT;
constexpr size_t OFF_WKT  = OFF_WKVT + SZ_WKVT;
constexpr size_t OFF_WVT  = OFF_WKT  + SZ_WKT;
constexpr size_t OFF_WOT  = OFF_WVT  + SZ_WVT;
constexpr size_t OFF_CB   = OFF_WOT  + SZ_WOT;
constexpr size_t OFF_QB   = OFF_CB   + SZ_CB;
constexpr size_t OFF_KB   = OFF_QB   + SZ_QB;
constexpr size_t OFF_VT   = OFF_KB   + SZ_KB;
constexpr size_t OFF_AB   = OFF_VT   + SZ_VT;

// ---------------- WMMA helpers ----------------
__device__ inline v8f wmma_bf16(v16bf a, v16bf b, v8f c) {
    return __builtin_amdgcn_wmma_f32_16x16x32_bf16(false, a, false, b,
                                                   (short)0, c, false, false);
}

// A fragment: 16x32 tile from row-major bf16; two contiguous 16B runs per lane.
__device__ inline v16bf frag_A(const bf16_t* s, int ld, int row0, int k0, int lane) {
    const bf16_t* p = s + (row0 + (lane & 15)) * ld + k0 + ((lane >> 4) << 3);
    union { v16bf v; v8bf h[2]; } u;
    u.h[0] = *(const v8bf*)(p);
    u.h[1] = *(const v8bf*)(p + 16);
    return u.v;
}

// B fragment (32x16 KxN) from N-major storage s[n*ld + k].
__device__ inline v16bf frag_B(const bf16_t* s, int ld, int n0, int k0, int lane) {
    const bf16_t* p = s + (n0 + (lane & 15)) * ld + k0 + ((lane >> 4) << 3);
    union { v16bf v; v8bf h[2]; } u;
    u.h[0] = *(const v8bf*)(p);
    u.h[1] = *(const v8bf*)(p + 16);
    return u.v;
}

__device__ inline v4bf cvt4(v4f x) {
    v4bf r;
    r[0] = (bf16_t)x[0]; r[1] = (bf16_t)x[1];
    r[2] = (bf16_t)x[2]; r[3] = (bf16_t)x[3];
    return r;
}

// ---------------- f32 -> bf16 copy (vectorized) ----------------
__global__ __launch_bounds__(256)
void mla_cvt_bf16(const float* __restrict__ in, bf16_t* __restrict__ out)
{
    const size_t i = ((size_t)blockIdx.x * 256 + threadIdx.x) * 4;
    *(v4bf*)(out + i) = cvt4(*(const v4f*)(in + i));
}

// ---------------- W[K,N] f32 -> Wt[N,K] bf16 (LDS-tiled transpose) ----------------
__global__ __launch_bounds__(256)
void mla_transpose_cvt(const float* __restrict__ in, bf16_t* __restrict__ out,
                       int K, int N)
{
    __shared__ float t[32][33];
    const int k0 = blockIdx.y * 32, n0 = blockIdx.x * 32;
    const int tx = threadIdx.x, ty = threadIdx.y;   // (32, 8)
#pragma unroll
    for (int i = 0; i < 4; ++i)
        t[ty + 8 * i][tx] = in[(size_t)(k0 + ty + 8 * i) * N + n0 + tx];
    __syncthreads();
#pragma unroll
    for (int i = 0; i < 4; ++i)
        out[(size_t)(n0 + ty + 8 * i) * K + k0 + tx] = (bf16_t)t[tx][ty + 8 * i];
}

// ---------------- GEMM + bias:  C[M,N] = A[M,K] @ Wt[N,K]^T + bias[N] -----------
// Double-buffered async LDS staging; 8 waves, tile 128x128, BK=32.
constexpr int GBM = 128, GBN = 128, GBK = 32;
constexpr int GLD = GBK + 8;   // 40 elements (80B rows, 16B aligned)

__global__ __launch_bounds__(256)
void mla_gemm_bias(const bf16_t* __restrict__ A, const bf16_t* __restrict__ Wt,
                   const float* __restrict__ bias, float* __restrict__ C,
                   int M, int N, int K)
{
    __shared__ alignas(16) bf16_t As[2][GBM * GLD];
    __shared__ alignas(16) bf16_t Bs[2][GBN * GLD];

    const int tid  = threadIdx.x;
    const int wave = tid >> 5;
    const int lane = tid & 31;
    const int half = lane >> 4;
    const int l16  = lane & 15;
    const int wr   = wave & 3;       // 4 row-groups of 32
    const int wc   = wave >> 2;      // 2 col-groups of 64
    const int bm   = blockIdx.y * GBM;
    const int bn   = blockIdx.x * GBN;

    auto stage = [&](int k0, int buf) {
#pragma unroll
        for (int i = 0; i < 2; ++i) {
            int e = tid + i * 256;
            int r = e >> 2, c8 = e & 3;
            async_cp16(A + (size_t)(bm + r) * K + k0 + c8 * 8,
                       As[buf] + r * GLD + c8 * 8);
        }
#pragma unroll
        for (int i = 0; i < 2; ++i) {
            int e = tid + i * 256;
            int n = e >> 2, c8 = e & 3;
            async_cp16(Wt + (size_t)(bn + n) * K + k0 + c8 * 8,
                       Bs[buf] + n * GLD + c8 * 8);
        }
    };

    v8f zero = {};
    v8f acc[2][4];
#pragma unroll
    for (int mi = 0; mi < 2; ++mi)
#pragma unroll
        for (int t = 0; t < 4; ++t) acc[mi][t] = zero;

    stage(0, 0);
    wait_async();
    __syncthreads();

    int cur = 0;
    for (int k0 = 0; k0 < K; k0 += GBK) {
        if (k0 + GBK < K) stage(k0 + GBK, cur ^ 1);   // overlap with WMMAs below

        v16bf a0 = frag_A(As[cur], GLD, wr * 32,      0, lane);
        v16bf a1 = frag_A(As[cur], GLD, wr * 32 + 16, 0, lane);
#pragma unroll
        for (int t = 0; t < 4; ++t) {
            v16bf bb = frag_B(Bs[cur], GLD, wc * 64 + t * 16, 0, lane);
            acc[0][t] = wmma_bf16(a0, bb, acc[0][t]);
            acc[1][t] = wmma_bf16(a1, bb, acc[1][t]);
        }
        wait_async();
        __syncthreads();
        cur ^= 1;
    }

    const size_t Ns = (size_t)N;
#pragma unroll
    for (int mi = 0; mi < 2; ++mi) {
        const int rbase = bm + wr * 32 + mi * 16 + half * 8;
#pragma unroll
        for (int t = 0; t < 4; ++t) {
            const int n    = bn + wc * 64 + t * 16 + l16;
            const float bv = bias[n];
            float* cp = C + (size_t)rbase * Ns + n;
#pragma unroll
            for (int r = 0; r < 8; ++r)
                cp[(size_t)r * Ns] = acc[mi][t][r] + bv;
        }
    }
}

// ---------------- LayerNorm over the 512-wide c slice of kv -> bf16 ----------------
__global__ __launch_bounds__(256)
void mla_layernorm(const float* __restrict__ kv, const float* __restrict__ g,
                   const float* __restrict__ bta, bf16_t* __restrict__ out)
{
    const int row = blockIdx.x;
    const int tid = threadIdx.x;
    const float* x = kv + (size_t)row * (KV_RANK_ + H_*ROPE_) + H_*ROPE_;

    __shared__ float ssum[256];
    __shared__ float ssq[256];
    float v0 = x[tid], v1 = x[tid + 256];
    ssum[tid] = v0 + v1;
    ssq[tid]  = v0 * v0 + v1 * v1;
    __syncthreads();
    for (int off = 128; off > 0; off >>= 1) {
        if (tid < off) { ssum[tid] += ssum[tid + off]; ssq[tid] += ssq[tid + off]; }
        __syncthreads();
    }
    const float mean = ssum[0] * (1.0f / 512.0f);
    const float var  = ssq[0] * (1.0f / 512.0f) - mean * mean;
    const float inv  = rsqrtf(var + 1e-5f);
    bf16_t* o = out + (size_t)row * KV_RANK_;
    o[tid]       = (bf16_t)((v0 - mean) * inv * g[tid]       + bta[tid]);
    o[tid + 256] = (bf16_t)((v1 - mean) * inv * g[tid + 256] + bta[tid + 256]);
}

// ---------------- fused RoPE + pack of bf16 Q/K caches [B,S,H,192] ----------------
__global__ __launch_bounds__(256)
void mla_rope_pack(const float* __restrict__ qf, const float* __restrict__ knf,
                   const float* __restrict__ kvf,
                   const float* __restrict__ cosT, const float* __restrict__ sinT,
                   bf16_t* __restrict__ q_bf, bf16_t* __restrict__ k_bf)
{
    const int idx = blockIdx.x * 256 + threadIdx.x;   // B*S*H*96 threads
    const int i    = idx % 96;
    const int rest = idx / 96;
    const int h    = rest & 15;
    const int s    = (rest >> 4) & (S_ - 1);
    const int b    = rest >> 15;
    const int d    = 2 * i;
    const size_t bs  = (size_t)(b * S_ + s);
    const size_t off = bs * (H_*QKHD_) + h * QKHD_ + d;

    float qa, qb2, ka, kb2;
    if (d < NOPE_) {
        qa = qf[off]; qb2 = qf[off + 1];
        size_t kn = bs * (H_*NOPE_) + h * NOPE_ + d;
        ka = knf[kn]; kb2 = knf[kn + 1];
    } else {
        int pair = (d - NOPE_) >> 1;
        float c = cosT[s * (ROPE_/2) + pair], sn = sinT[s * (ROPE_/2) + pair];
        float a0 = qf[off], b0 = qf[off + 1];
        qa  = a0 * c - b0 * sn;
        qb2 = a0 * sn + b0 * c;
        size_t kr = bs * (KV_RANK_ + H_*ROPE_) + h * ROPE_ + (d - NOPE_);
        float a1 = kvf[kr], b1 = kvf[kr + 1];
        ka  = a1 * c - b1 * sn;
        kb2 = a1 * sn + b1 * c;
    }
    q_bf[off]     = (bf16_t)qa;
    q_bf[off + 1] = (bf16_t)qb2;
    k_bf[off]     = (bf16_t)ka;
    k_bf[off + 1] = (bf16_t)kb2;
}

// ---------------- V pack: v f32 [B,S,H,128] -> vT bf16 [B,H,128,S] ----------------
__global__ __launch_bounds__(256)
void mla_vpack(const float* __restrict__ vf, bf16_t* __restrict__ vT)
{
    __shared__ float t[32][33];
    const int s0 = blockIdx.x * 32;
    const int d0 = blockIdx.y * 32;
    const int bh = blockIdx.z;          // b*H + h
    const int b  = bh >> 4, h = bh & 15;
    const int tx = threadIdx.x, ty = threadIdx.y;   // (32, 8)
#pragma unroll
    for (int i = 0; i < 4; ++i)
        t[ty + 8 * i][tx] = vf[(size_t)(b * S_ + s0 + ty + 8 * i) * (H_*VHD_)
                               + h * VHD_ + d0 + tx];
    __syncthreads();
#pragma unroll
    for (int i = 0; i < 4; ++i)
        vT[((size_t)bh * VHD_ + d0 + ty + 8 * i) * S_ + s0 + tx] =
            (bf16_t)t[tx][ty + 8 * i];
}

// ---------------- Flash-style causal attention (bf16 caches) ----------------
// Block = (b, h, 64-row query tile); 4 waves, wave owns 16 query rows.
// Double-buffered async K/V staging overlapped with QK^T/softmax/PV.
constexpr int TQ = 64, TK = 64;
constexpr int LDQK = QKHD_ + 8;   // 200 elems
constexpr int LDVT = TK + 8;      // 72 elems
constexpr int LDP  = 72;

__global__ __launch_bounds__(128)
void mla_attention(const bf16_t* __restrict__ qc, const bf16_t* __restrict__ kc,
                   const bf16_t* __restrict__ vT, bf16_t* __restrict__ ab)
{
    __shared__ alignas(16) bf16_t Qs[TQ * LDQK];
    __shared__ alignas(16) bf16_t Ks[2][TK * LDQK];    // [key][qk-dim]
    __shared__ alignas(16) bf16_t Vs[2][VHD_ * LDVT];  // [v-dim][key]
    __shared__ alignas(16) bf16_t Ps[4][16 * LDP];

    const int tid  = threadIdx.x;
    const int lane = tid & 31;
    const int wave = tid >> 5;
    const int half = lane >> 4;
    const int l16  = lane & 15;
    const int it   = blockIdx.x;        // query tile
    const int h    = blockIdx.y;
    const int b    = blockIdx.z;
    const int q0   = it * TQ;
    const float scale = 0.07216878364870323f;   // 192^-0.5

    auto stage_kv = [&](int j, int buf) {
        const int k0 = j * TK;
#pragma unroll
        for (int i = 0; i < 12; ++i) {
            int e = tid + i * 128;
            int r = e / 24, c8 = e % 24;
            async_cp16(kc + (size_t)(b * S_ + k0 + r) * (H_*QKHD_)
                          + h * QKHD_ + c8 * 8,
                       Ks[buf] + r * LDQK + c8 * 8);
        }
#pragma unroll
        for (int i = 0; i < 8; ++i) {
            int e = tid + i * 128;
            int d = e >> 3, c8 = e & 7;
            async_cp16(vT + ((size_t)(b * H_ + h) * VHD_ + d) * S_ + k0 + c8 * 8,
                       Vs[buf] + d * LDVT + c8 * 8);
        }
    };

    // Q tile (async) + first K/V tile
#pragma unroll
    for (int i = 0; i < 12; ++i) {
        int e = tid + i * 128;
        int r = e / 24, c8 = e % 24;
        async_cp16(qc + (size_t)(b * S_ + q0 + r) * (H_*QKHD_) + h * QKHD_ + c8 * 8,
                   Qs + r * LDQK + c8 * 8);
    }
    stage_kv(0, 0);
    wait_async();
    __syncthreads();

    v8f zero = {};
    v8f o[8];
#pragma unroll
    for (int t = 0; t < 8; ++t) o[t] = zero;
    float m_i[8], l_i[8];
#pragma unroll
    for (int r = 0; r < 8; ++r) { m_i[r] = -INFINITY; l_i[r] = 0.0f; }

    const int qr = wave * 16;           // this wave's query rows within tile

    int cur = 0;
    for (int j = 0; j <= it; ++j) {
        if (j < it) stage_kv(j + 1, cur ^ 1);   // overlap with compute below

        // S = Q K^T  (16 x 64 per wave)
        v8f sc[4];
#pragma unroll
        for (int t = 0; t < 4; ++t) sc[t] = zero;
#pragma unroll
        for (int ks = 0; ks < QKHD_ / 32; ++ks) {       // 6
            v16bf a = frag_A(Qs, LDQK, qr, ks * 32, lane);
#pragma unroll
            for (int t = 0; t < 4; ++t) {
                v16bf bb = frag_B(Ks[cur], LDQK, t * 16, ks * 32, lane);
                sc[t] = wmma_bf16(a, bb, sc[t]);
            }
        }
        // scale + causal mask (diagonal tile only)
#pragma unroll
        for (int t = 0; t < 4; ++t)
#pragma unroll
            for (int r = 0; r < 8; ++r) sc[t][r] *= scale;
        if (j == it) {
#pragma unroll
            for (int t = 0; t < 4; ++t)
#pragma unroll
                for (int r = 0; r < 8; ++r) {
                    int qi = qr + r + half * 8;
                    int ki = t * 16 + l16;
                    if (ki > qi) sc[t][r] = -1e30f;
                }
        }
        // online softmax per row (rows live across the 16-lane half-wave)
        float alpha[8];
#pragma unroll
        for (int r = 0; r < 8; ++r) {
            float mx = fmaxf(fmaxf(sc[0][r], sc[1][r]), fmaxf(sc[2][r], sc[3][r]));
            mx = fmaxf(mx, __shfl_xor(mx, 1, 32));
            mx = fmaxf(mx, __shfl_xor(mx, 2, 32));
            mx = fmaxf(mx, __shfl_xor(mx, 4, 32));
            mx = fmaxf(mx, __shfl_xor(mx, 8, 32));
            float mn = fmaxf(m_i[r], mx);
            alpha[r] = __expf(m_i[r] - mn);
            m_i[r] = mn;
            float rs = 0.0f;
#pragma unroll
            for (int t = 0; t < 4; ++t) {
                float p = __expf(sc[t][r] - mn);
                sc[t][r] = p;
                rs += p;
            }
            rs += __shfl_xor(rs, 1, 32);
            rs += __shfl_xor(rs, 2, 32);
            rs += __shfl_xor(rs, 4, 32);
            rs += __shfl_xor(rs, 8, 32);
            l_i[r] = l_i[r] * alpha[r] + rs;
        }
        // rescale O
#pragma unroll
        for (int t = 0; t < 8; ++t)
#pragma unroll
            for (int r = 0; r < 8; ++r) o[t][r] *= alpha[r];
        // P: C-layout -> A-layout via per-wave LDS patch (same-wave, in-order DS)
#pragma unroll
        for (int t = 0; t < 4; ++t)
#pragma unroll
            for (int r = 0; r < 8; ++r)
                Ps[wave][(r + half * 8) * LDP + t * 16 + l16] = (bf16_t)sc[t][r];
        // O += P V  (K = 64 -> two WMMA steps)
#pragma unroll
        for (int ks = 0; ks < 2; ++ks) {
            v16bf a = frag_A(Ps[wave], LDP, 0, ks * 32, lane);
#pragma unroll
            for (int t = 0; t < 8; ++t) {
                v16bf bb = frag_B(Vs[cur], LDVT, t * 16, ks * 32, lane);
                o[t] = wmma_bf16(a, bb, o[t]);
            }
        }
        wait_async();
        __syncthreads();
        cur ^= 1;
    }

    // normalize and store (bf16, consumed only as GEMM-A by out-proj)
#pragma unroll
    for (int r = 0; r < 8; ++r) {
        float inv = 1.0f / l_i[r];
#pragma unroll
        for (int t = 0; t < 8; ++t) o[t][r] *= inv;
    }
    const int qrow = q0 + qr + half * 8;
#pragma unroll
    for (int t = 0; t < 8; ++t) {
        int col = h * VHD_ + t * 16 + l16;
        bf16_t* op = ab + (size_t)(b * S_ + qrow) * (H_*VHD_) + col;
#pragma unroll
        for (int r = 0; r < 8; ++r)
            op[(size_t)r * (H_*VHD_)] = (bf16_t)o[t][r];
    }
}

// ---------------- launcher ----------------
extern "C" void kernel_launch(void* const* d_in, const int* in_sizes, int n_in,
                              void* d_out, int out_size, void* d_ws, size_t ws_size,
                              hipStream_t stream)
{
    const float* x      = (const float*)d_in[0];
    const float* wq_w   = (const float*)d_in[1];
    const float* wq_b   = (const float*)d_in[2];
    const float* wkv_a_w= (const float*)d_in[3];
    const float* wkv_a_b= (const float*)d_in[4];
    const float* kvn_g  = (const float*)d_in[5];
    const float* kvn_b  = (const float*)d_in[6];
    const float* wk_w   = (const float*)d_in[7];
    const float* wk_b   = (const float*)d_in[8];
    const float* wv_w   = (const float*)d_in[9];
    const float* wv_b   = (const float*)d_in[10];
    const float* wo_w   = (const float*)d_in[11];
    const float* wo_b   = (const float*)d_in[12];
    const float* cosT   = (const float*)d_in[13];
    const float* sinT   = (const float*)d_in[14];
    // d_in[15] = mask (bool): causal, handled analytically in the attention kernel

    char* ws = (char*)d_ws;
    float*  q_f   = (float*)(ws + OFF_QF);
    float*  kv_f  = (float*)(ws + OFF_KVF);
    float*  kn_f  = (float*)(ws + OFF_KNF);
    float*  v_f   = (float*)(ws + OFF_VF);
    bf16_t* x_bf  = (bf16_t*)(ws + OFF_XB);
    bf16_t* wq_t  = (bf16_t*)(ws + OFF_WQT);
    bf16_t* wkv_t = (bf16_t*)(ws + OFF_WKVT);
    bf16_t* wk_t  = (bf16_t*)(ws + OFF_WKT);
    bf16_t* wv_t  = (bf16_t*)(ws + OFF_WVT);
    bf16_t* wo_t  = (bf16_t*)(ws + OFF_WOT);
    bf16_t* c_bf  = (bf16_t*)(ws + OFF_CB);
    bf16_t* q_bf  = (bf16_t*)(ws + OFF_QB);
    bf16_t* k_bf  = (bf16_t*)(ws + OFF_KB);
    bf16_t* vT_bf = (bf16_t*)(ws + OFF_VT);
    bf16_t* a_bf  = (bf16_t*)(ws + OFF_AB);
    float*  out   = (float*)d_out;

    dim3 blk(256);
    dim3 tblk(32, 8);

    // one-time bf16 packing of operands (fragment-friendly layouts)
    mla_cvt_bf16<<<(M_*DIM_)/1024, blk, 0, stream>>>(x, x_bf);
    mla_transpose_cvt<<<dim3((H_*QKHD_)/32, DIM_/32), tblk, 0, stream>>>(
        wq_w, wq_t, DIM_, H_*QKHD_);
    mla_transpose_cvt<<<dim3((KV_RANK_+H_*ROPE_)/32, DIM_/32), tblk, 0, stream>>>(
        wkv_a_w, wkv_t, DIM_, KV_RANK_+H_*ROPE_);
    mla_transpose_cvt<<<dim3((H_*NOPE_)/32, KV_RANK_/32), tblk, 0, stream>>>(
        wk_w, wk_t, KV_RANK_, H_*NOPE_);
    mla_transpose_cvt<<<dim3((H_*VHD_)/32, KV_RANK_/32), tblk, 0, stream>>>(
        wv_w, wv_t, KV_RANK_, H_*VHD_);
    mla_transpose_cvt<<<dim3(DIM_/32, (H_*VHD_)/32), tblk, 0, stream>>>(
        wo_w, wo_t, H_*VHD_, DIM_);

    // q = x @ wq_w + wq_b           [4096,3072] f32
    mla_gemm_bias<<<dim3((H_*QKHD_)/GBN, M_/GBM), blk, 0, stream>>>(
        x_bf, wq_t, wq_b, q_f, M_, H_*QKHD_, DIM_);
    // kv = x @ wkv_a_w + wkv_a_b    [4096,1536] f32
    mla_gemm_bias<<<dim3((KV_RANK_+H_*ROPE_)/GBN, M_/GBM), blk, 0, stream>>>(
        x_bf, wkv_t, wkv_a_b, kv_f, M_, KV_RANK_+H_*ROPE_, DIM_);
    // c = LayerNorm(kv[..., 1024:]) -> bf16
    mla_layernorm<<<M_, 256, 0, stream>>>(kv_f, kvn_g, kvn_b, c_bf);
    // k_nope = c @ wk_w + wk_b      [4096,2048] f32
    mla_gemm_bias<<<dim3((H_*NOPE_)/GBN, M_/GBM), blk, 0, stream>>>(
        c_bf, wk_t, wk_b, kn_f, M_, H_*NOPE_, KV_RANK_);
    // v = c @ wv_w + wv_b           [4096,2048] f32
    mla_gemm_bias<<<dim3((H_*VHD_)/GBN, M_/GBM), blk, 0, stream>>>(
        c_bf, wv_t, wv_b, v_f, M_, H_*VHD_, KV_RANK_);
    // bf16 Q/K caches with fused RoPE; transposed bf16 V cache
    mla_rope_pack<<<(B_*S_*H_*96)/256, blk, 0, stream>>>(
        q_f, kn_f, kv_f, cosT, sinT, q_bf, k_bf);
    mla_vpack<<<dim3(S_/32, VHD_/32, B_*H_), tblk, 0, stream>>>(v_f, vT_bf);
    // causal attention -> a_bf [4096, 2048] bf16
    mla_attention<<<dim3(S_/TQ, H_, B_), 128, 0, stream>>>(
        q_bf, k_bf, vT_bf, a_bf);
    // out = attn @ wo_w + wo_b      [4096,2048] f32
    mla_gemm_bias<<<dim3(DIM_/GBN, M_/GBM), blk, 0, stream>>>(
        a_bf, wo_t, wo_b, out, M_, DIM_, H_*VHD_);
}
